// MultiHeadAttention_7138235646640
// MI455X (gfx1250) — matrix operand order repaired and checked by reading, not compiled, not run on hardware
//
#include <hip/hip_runtime.h>

typedef __bf16 bf16;
typedef __attribute__((ext_vector_type(16))) __bf16 v16bf;
typedef __attribute__((ext_vector_type(8)))  __bf16 v8bf;
typedef __attribute__((ext_vector_type(8)))  float  v8f;

#define B_   4
#define T_   2048
#define DIN  1024
#define DOUT 1024
#define NH   16
#define HD   64

// ---------------------------------------------------------------------------
// WMMA helper: D = A(16x32 bf16) * B(32x16 bf16) + C(16x16 f32)
// ---------------------------------------------------------------------------
__device__ __forceinline__ v8f wmma_bf16(v16bf a, v16bf b, v8f c) {
  return __builtin_amdgcn_wmma_f32_16x16x32_bf16(false, a, false, b, (short)0, c,
                                                 false, false);
}

// CDNA5 async copy: 16 bytes/lane, global -> LDS, tracked by ASYNCcnt.
__device__ __forceinline__ void async_copy16(unsigned lds_off, const void* gaddr) {
  asm volatile("global_load_async_to_lds_b128 %0, %1, off"
               :: "v"(lds_off), "v"(gaddr)
               : "memory");
}
__device__ __forceinline__ void wait_async0() {
  asm volatile("s_wait_asynccnt 0" ::: "memory");
}

// A-fragment (16x32, MxK), row-major source (global or LDS), 16B-aligned rows.
__device__ __forceinline__ v16bf load_fragA(const bf16* __restrict__ base, int ld) {
  const int lane = threadIdx.x & 31;
  const int m    = lane & 15;
  const int koff = (lane >> 4) << 3;   // 0 or 8
  const bf16* p = base + (size_t)m * ld + koff;
  v8bf lo = *(const v8bf*)(p);
  v8bf hi = *(const v8bf*)(p + 16);
  v16bf r;
#pragma unroll
  for (int i = 0; i < 8; ++i) { r[i] = lo[i]; r[i + 8] = hi[i]; }
  return r;
}

// B-fragment (32x16, KxN). Source is "n-major": row n holds B[k][n] contiguous
// in k. Lanes 0-15 hold K=0..7,16..23; lanes 16-31 hold K=8..15,24..31.
__device__ __forceinline__ v16bf load_fragB(const bf16* __restrict__ base, int ld) {
  const int lane = threadIdx.x & 31;
  const int n    = lane & 15;
  const int koff = (lane >> 4) << 3;   // 0 or 8
  const bf16* p = base + (size_t)n * ld + koff;
  v8bf lo = *(const v8bf*)(p);
  v8bf hi = *(const v8bf*)(p + 16);
  v16bf r;
#pragma unroll
  for (int i = 0; i < 8; ++i) { r[i] = lo[i]; r[i + 8] = hi[i]; }
  return r;
}

// ---------------------------------------------------------------------------
// fp32 -> bf16 cast
// ---------------------------------------------------------------------------
__global__ void cast_f32_bf16(const float* __restrict__ src, bf16* __restrict__ dst,
                              int n) {
  int i = blockIdx.x * blockDim.x + threadIdx.x;
  int stride = gridDim.x * blockDim.x;
  for (; i < n; i += stride) dst[i] = (bf16)src[i];
}

// ---------------------------------------------------------------------------
// QKV projection: y = x @ W^T, 32x32 output tile per wave (2x2 WMMA blocking).
// Q,K stored [B,H,T,hd] bf16; V stored transposed [B,H,hd,T] bf16.
// ---------------------------------------------------------------------------
__global__ void __launch_bounds__(128) qkv_proj_kernel(
    const bf16* __restrict__ xbf, const bf16* __restrict__ Wq,
    const bf16* __restrict__ Wk, const bf16* __restrict__ Wv,
    bf16* __restrict__ Q, bf16* __restrict__ K, bf16* __restrict__ VT) {
  const int b     = blockIdx.z % B_;
  const int which = blockIdx.z / B_;
  const int t0 = blockIdx.x * 32;
  const int o0 = (blockIdx.y * blockDim.y + threadIdx.y) * 32;
  const bf16* W = (which == 0) ? Wq : (which == 1) ? Wk : Wv;
  const bf16* xrow = xbf + ((size_t)b * T_ + t0) * DIN;

  v8f acc00 = {}, acc01 = {}, acc10 = {}, acc11 = {};
#pragma unroll 2
  for (int k0 = 0; k0 < DIN; k0 += 32) {
    v16bf a0 = load_fragA(xrow + k0, DIN);
    v16bf a1 = load_fragA(xrow + (size_t)16 * DIN + k0, DIN);
    v16bf b0 = load_fragB(W + (size_t)o0 * DIN + k0, DIN);
    v16bf b1 = load_fragB(W + (size_t)(o0 + 16) * DIN + k0, DIN);
    acc00 = wmma_bf16(a0, b0, acc00);
    acc01 = wmma_bf16(a0, b1, acc01);
    acc10 = wmma_bf16(a1, b0, acc10);
    acc11 = wmma_bf16(a1, b1, acc11);
  }

  const int lane = threadIdx.x & 31;
  const int n = lane & 15, hsel = lane >> 4;
#pragma unroll
  for (int mi = 0; mi < 2; ++mi) {
#pragma unroll
    for (int ni = 0; ni < 2; ++ni) {
      const v8f& acc = (mi == 0) ? (ni == 0 ? acc00 : acc01)
                                 : (ni == 0 ? acc10 : acc11);
      const int o = o0 + ni * 16 + n;
      const int h = o / HD, d = o % HD;
#pragma unroll
      for (int r = 0; r < 8; ++r) {
        const int t = t0 + mi * 16 + r + 8 * hsel;
        const bf16 v = (bf16)acc[r];
        if (which == 0)
          Q[((size_t)(b * NH + h) * T_ + t) * HD + d] = v;
        else if (which == 1)
          K[((size_t)(b * NH + h) * T_ + t) * HD + d] = v;
        else
          VT[((size_t)(b * NH + h) * HD + d) * T_ + t] = v;
      }
    }
  }
}

// ---------------------------------------------------------------------------
// Flash attention: one wave per (b, h, 16-row q-tile). kv tiles of 32, with
// K/V tiles double-buffered in LDS via async global->LDS copies (ASYNCcnt).
// ---------------------------------------------------------------------------
__global__ void __launch_bounds__(32) attn_kernel(const bf16* __restrict__ Q,
                                                  const bf16* __restrict__ K,
                                                  const bf16* __restrict__ VT,
                                                  bf16* __restrict__ ctx) {
  __shared__ __align__(16) bf16 kbuf[2][32 * HD];   // K tile: 32 kv x 64 d (4KB each)
  __shared__ __align__(16) bf16 vbuf[2][HD * 32];   // V^T tile: 64 d x 32 kv (4KB each)
  __shared__ __align__(16) float pbuf[16][32];      // P transpose staging (2KB)

  const int b = blockIdx.z, h = blockIdx.y;
  const int t0 = blockIdx.x * 16;
  const bf16* Qbh = Q + (size_t)(b * NH + h) * T_ * HD;
  const bf16* Kbh = K + (size_t)(b * NH + h) * T_ * HD;
  const bf16* Vbh = VT + (size_t)(b * NH + h) * HD * T_;
  const int lane = threadIdx.x & 31;
  const int n = lane & 15, hsel = lane >> 4;

  // stage K tile (32 rows x 128B, linear 4KB) + V^T tile (64 rows x 64B strided)
  auto stage_kv = [&](int c0, int buf) {
    const char* gk = (const char*)(Kbh + (size_t)c0 * HD);
    const unsigned lk = (unsigned)(size_t)&kbuf[buf][0];
#pragma unroll
    for (int j = 0; j < 8; ++j) {
      const unsigned off = (unsigned)(j * 32 + lane) * 16u;
      async_copy16(lk + off, gk + off);
    }
    const char* gv = (const char*)(Vbh + c0);
    const unsigned lv = (unsigned)(size_t)&vbuf[buf][0];
    const int cb = (lane & 3) * 16;            // 4 lanes cover one 64B row seg
#pragma unroll
    for (int j = 0; j < 8; ++j) {
      const int row = j * 8 + (lane >> 2);
      async_copy16(lv + (unsigned)(row * 64 + cb),
                   gv + (size_t)row * (T_ * sizeof(bf16)) + cb);
    }
  };

  const v16bf qa0 = load_fragA(Qbh + (size_t)t0 * HD, HD);
  const v16bf qa1 = load_fragA(Qbh + (size_t)t0 * HD + 32, HD);

  float mrow[8], lrow[8];
  v8f oa0 = {}, oa1 = {}, oa2 = {}, oa3 = {};
#pragma unroll
  for (int r = 0; r < 8; ++r) { mrow[r] = -__builtin_inff(); lrow[r] = 0.f; }

  int buf = 0;
  stage_kv(0, 0);
  wait_async0();

  for (int c0 = 0; c0 <= t0 + 15; c0 += 32) {   // causal: skip upper triangle
    if (c0 + 32 <= t0 + 15) stage_kv(c0 + 32, buf ^ 1);   // prefetch next tile

    v8f s0 = {}, s1 = {};
    v16bf kb;
    kb = load_fragB(&kbuf[buf][0], HD);              // kv c0+n,  k 0..31
    s0 = wmma_bf16(qa0, kb, s0);
    kb = load_fragB(&kbuf[buf][0] + 32, HD);         // kv c0+n,  k 32..63
    s0 = wmma_bf16(qa1, kb, s0);
    kb = load_fragB(&kbuf[buf][16 * HD], HD);        // kv c0+16+n
    s1 = wmma_bf16(qa0, kb, s1);
    kb = load_fragB(&kbuf[buf][16 * HD] + 32, HD);
    s1 = wmma_bf16(qa1, kb, s1);

    float p0[8], p1[8], alpha[8];
#pragma unroll
    for (int r = 0; r < 8; ++r) {
      const int q = t0 + r + 8 * hsel;
      float v0 = (c0 + n      <= q) ? s0[r] * 0.125f : -__builtin_inff();
      float v1 = (c0 + 16 + n <= q) ? s1[r] * 0.125f : -__builtin_inff();
      float mt = fmaxf(v0, v1);
#pragma unroll
      for (int msk = 1; msk <= 8; msk <<= 1)
        mt = fmaxf(mt, __shfl_xor(mt, msk, 32));     // stays within 16-lane half
      const float mnew = fmaxf(mrow[r], mt);
      const float a  = __expf(mrow[r] - mnew);       // exp(-inf)=0 on first tile
      const float e0 = __expf(v0 - mnew);
      const float e1 = __expf(v1 - mnew);
      float lt = e0 + e1;
#pragma unroll
      for (int msk = 1; msk <= 8; msk <<= 1)
        lt += __shfl_xor(lt, msk, 32);
      lrow[r] = lrow[r] * a + lt;
      mrow[r] = mnew;
      alpha[r] = a;
      p0[r] = e0; p1[r] = e1;
    }

#pragma unroll
    for (int r = 0; r < 8; ++r) {
      oa0[r] *= alpha[r]; oa1[r] *= alpha[r];
      oa2[r] *= alpha[r]; oa3[r] *= alpha[r];
    }

    __syncthreads();
#pragma unroll
    for (int r = 0; r < 8; ++r) {              // C-layout -> LDS 16x32 tile
      pbuf[r + 8 * hsel][n]      = p0[r];
      pbuf[r + 8 * hsel][16 + n] = p1[r];
    }
    __syncthreads();

    v16bf pa;                                   // re-read in A-fragment layout
    {
      const int m = lane & 15;
      const int koff = (lane >> 4) << 3;
#pragma unroll
      for (int i = 0; i < 8; ++i) {
        pa[i]     = (bf16)pbuf[m][koff + i];
        pa[i + 8] = (bf16)pbuf[m][16 + koff + i];
      }
    }
    v16bf vb;
    vb = load_fragB(&vbuf[buf][0 * 32],  32); oa0 = wmma_bf16(pa, vb, oa0);
    vb = load_fragB(&vbuf[buf][16 * 32], 32); oa1 = wmma_bf16(pa, vb, oa1);
    vb = load_fragB(&vbuf[buf][32 * 32], 32); oa2 = wmma_bf16(pa, vb, oa2);
    vb = load_fragB(&vbuf[buf][48 * 32], 32); oa3 = wmma_bf16(pa, vb, oa3);
    __syncthreads();

    wait_async0();                              // next KV tile has landed
    buf ^= 1;
  }

#pragma unroll
  for (int r = 0; r < 8; ++r) {
    const float inv = 1.0f / lrow[r];
    const int t = t0 + r + 8 * hsel;
    bf16* crow = ctx + ((size_t)b * T_ + t) * DOUT + h * HD;
    crow[n]      = (bf16)(oa0[r] * inv);
    crow[16 + n] = (bf16)(oa1[r] * inv);
    crow[32 + n] = (bf16)(oa2[r] * inv);
    crow[48 + n] = (bf16)(oa3[r] * inv);
  }
}

// ---------------------------------------------------------------------------
// Output projection: out = ctx @ Wo^T + bo (fp32 out), 32x32 tile per wave.
// ---------------------------------------------------------------------------
__global__ void __launch_bounds__(128) proj_kernel(const bf16* __restrict__ ctx,
                                                   const bf16* __restrict__ Wo,
                                                   const float* __restrict__ bo,
                                                   float* __restrict__ out) {
  const int row0 = blockIdx.x * 32;   // over B*T rows
  const int o0   = (blockIdx.y * blockDim.y + threadIdx.y) * 32;

  v8f acc00 = {}, acc01 = {}, acc10 = {}, acc11 = {};
#pragma unroll 2
  for (int k0 = 0; k0 < DOUT; k0 += 32) {
    v16bf a0 = load_fragA(ctx + (size_t)row0 * DOUT + k0, DOUT);
    v16bf a1 = load_fragA(ctx + (size_t)(row0 + 16) * DOUT + k0, DOUT);
    v16bf b0 = load_fragB(Wo + (size_t)o0 * DOUT + k0, DOUT);
    v16bf b1 = load_fragB(Wo + (size_t)(o0 + 16) * DOUT + k0, DOUT);
    acc00 = wmma_bf16(a0, b0, acc00);
    acc01 = wmma_bf16(a0, b1, acc01);
    acc10 = wmma_bf16(a1, b0, acc10);
    acc11 = wmma_bf16(a1, b1, acc11);
  }

  const int lane = threadIdx.x & 31;
  const int n = lane & 15, hsel = lane >> 4;
#pragma unroll
  for (int mi = 0; mi < 2; ++mi) {
#pragma unroll
    for (int ni = 0; ni < 2; ++ni) {
      const v8f& acc = (mi == 0) ? (ni == 0 ? acc00 : acc01)
                                 : (ni == 0 ? acc10 : acc11);
      const float bias = bo[o0 + ni * 16 + n];
#pragma unroll
      for (int r = 0; r < 8; ++r)
        out[(size_t)(row0 + mi * 16 + r + 8 * hsel) * DOUT + o0 + ni * 16 + n] =
            acc[r] + bias;
    }
  }
}

// ---------------------------------------------------------------------------
extern "C" void kernel_launch(void* const* d_in, const int* in_sizes, int n_in,
                              void* d_out, int out_size, void* d_ws, size_t ws_size,
                              hipStream_t stream) {
  const float* x  = (const float*)d_in[0];
  const float* Wq = (const float*)d_in[1];
  const float* Wk = (const float*)d_in[2];
  const float* Wv = (const float*)d_in[3];
  const float* Wo = (const float*)d_in[4];
  const float* bo = (const float*)d_in[5];
  float* out = (float*)d_out;

  char* ws = (char*)d_ws;
  size_t off = 0;
  auto alloc = [&](size_t bytes) -> char* {
    char* p = ws + off;
    off += (bytes + 255) & ~(size_t)255;
    return p;
  };
  bf16* xbf  = (bf16*)alloc((size_t)B_ * T_ * DIN * sizeof(bf16));
  bf16* Wqb  = (bf16*)alloc((size_t)DOUT * DIN * sizeof(bf16));
  bf16* Wkb  = (bf16*)alloc((size_t)DOUT * DIN * sizeof(bf16));
  bf16* Wvb  = (bf16*)alloc((size_t)DOUT * DIN * sizeof(bf16));
  bf16* Wob  = (bf16*)alloc((size_t)DOUT * DOUT * sizeof(bf16));
  bf16* Qb   = (bf16*)alloc((size_t)B_ * NH * T_ * HD * sizeof(bf16));
  bf16* Kb   = (bf16*)alloc((size_t)B_ * NH * T_ * HD * sizeof(bf16));
  bf16* VTb  = (bf16*)alloc((size_t)B_ * NH * HD * T_ * sizeof(bf16));
  bf16* ctxb = (bf16*)alloc((size_t)B_ * T_ * DOUT * sizeof(bf16));

  cast_f32_bf16<<<1024, 256, 0, stream>>>(x,  xbf, B_ * T_ * DIN);
  cast_f32_bf16<<<512,  256, 0, stream>>>(Wq, Wqb, DOUT * DIN);
  cast_f32_bf16<<<512,  256, 0, stream>>>(Wk, Wkb, DOUT * DIN);
  cast_f32_bf16<<<512,  256, 0, stream>>>(Wv, Wvb, DOUT * DIN);
  cast_f32_bf16<<<512,  256, 0, stream>>>(Wo, Wob, DOUT * DOUT);

  dim3 gblk(32, 4);
  qkv_proj_kernel<<<dim3(T_ / 32, DOUT / 128, B_ * 3), gblk, 0, stream>>>(
      xbf, Wqb, Wkb, Wvb, Qb, Kb, VTb);

  attn_kernel<<<dim3(T_ / 16, NH, B_), dim3(32), 0, stream>>>(Qb, Kb, VTb, ctxb);

  proj_kernel<<<dim3(B_ * T_ / 32, DOUT / 128), gblk, 0, stream>>>(ctxb, Wob, bo, out);
}
